// HgruReal2d_60997125538132
// MI455X (gfx1250) — matrix operand order repaired and checked by reading, not compiled, not run on hardware
//
#include <hip/hip_runtime.h>
#include <hip/hip_bf16.h>
#include <math.h>

// ---------------------------------------------------------------------------
// HgruReal2d for MI455X (gfx1250, wave32, WMMA)
//   H=64, W=64, B=16, D=256.  Rows = H*W*B = 65536.
//   GEMMs run on v_wmma_f32_16x16x32_bf16 (bf16 in, f32 accumulate),
//   software-pipelined: B double-buffered, A prefetched one K-chunk ahead.
// ---------------------------------------------------------------------------

typedef __attribute__((ext_vector_type(16))) __bf16          v16bf;
typedef __attribute__((ext_vector_type(16))) unsigned short  v16us;
typedef __attribute__((ext_vector_type(8)))  unsigned short  v8us;
typedef __attribute__((ext_vector_type(8)))  float           v8f;

#define H_DIM 64
#define W_DIM 64
#define B_DIM 16
#define D_DIM 256
#define ROWS  (H_DIM * W_DIM * B_DIM)              /* 65536 */
#define NTOT  ((size_t)ROWS * D_DIM)               /* 16,777,216 */
#define WPACK_ELEMS 65536                          /* per-weight packed bf16 count */

__device__ __forceinline__ unsigned short f2bf(float f) {
  unsigned u = __builtin_bit_cast(unsigned, f);
  u += 0x7FFFu + ((u >> 16) & 1u);                 // round-to-nearest-even
  return (unsigned short)(u >> 16);
}
__device__ __forceinline__ float sig_(float v) { return 1.0f / (1.0f + __expf(-v)); }

// ---------------------------------------------------------------------------
// Kernel 0a: pack fp32 weights [K=256][N=256] row-major -> bf16 in the exact
// per-lane B-operand order this code uses (contiguous k, same as A):
//   element t = ((kc*16 + nt)*32 + lane)*16 + j
//   lane <  16 : n = nt*16 + lane,      k = kc*32 + j
//   lane >= 16 : n = nt*16 + (lane-16), k = kc*32 + 16 + j
// ---------------------------------------------------------------------------
__global__ void k_pack_weights(const float* __restrict__ Wi, const float* __restrict__ Wl,
                               const float* __restrict__ Wg, const float* __restrict__ Wo,
                               unsigned short* __restrict__ wp) {
  int t = blockIdx.x * blockDim.x + threadIdx.x;   // 0 .. 4*65536-1
  int w = t >> 16;
  int e = t & 0xFFFF;
  int j    = e & 15;
  int lane = (e >> 4) & 31;
  int nt   = (e >> 9) & 15;
  int kc   = (e >> 13) & 7;
  int n = nt * 16 + (lane & 15);
  int k = kc * 32 + ((lane >> 4) ? 16 : 0) + j;
  const float* W = (w == 0) ? Wi : (w == 1) ? Wl : (w == 2) ? Wg : Wo;
  wp[t] = f2bf(W[k * D_DIM + n]);
}

// ---------------------------------------------------------------------------
// Kernel 0b: convert x (fp32) -> xbf (bf16), 8 elements per thread.
// ---------------------------------------------------------------------------
__global__ void k_cvt_x(const float* __restrict__ x, unsigned short* __restrict__ xbf) {
  size_t i = ((size_t)blockIdx.x * blockDim.x + threadIdx.x) * 8;
  float4 a = *reinterpret_cast<const float4*>(x + i);
  float4 b = *reinterpret_cast<const float4*>(x + i + 4);
  v8us o;
  o[0] = f2bf(a.x); o[1] = f2bf(a.y); o[2] = f2bf(a.z); o[3] = f2bf(a.w);
  o[4] = f2bf(b.x); o[5] = f2bf(b.y); o[6] = f2bf(b.z); o[7] = f2bf(b.w);
  *reinterpret_cast<v8us*>(xbf + i) = o;
}

// ---------------------------------------------------------------------------
// One wave computes a 16x256 output tile: acc[nt] = X[16 rows] @ W  (no bias;
// bias is folded into the caller's epilogue).  Flattened 128-step (kc,nt)
// loop, B double-buffered, A prefetched one K-chunk ahead.
// ---------------------------------------------------------------------------
__device__ __forceinline__ void wave_gemm16(const unsigned short* __restrict__ Xbf,
                                            const unsigned short* __restrict__ WPslot,
                                            int m0, int lane, v8f acc[16]) {
#pragma unroll
  for (int nt = 0; nt < 16; ++nt) acc[nt] = (v8f)(0.0f);

  const int row  = m0 * 16 + (lane & 15);
  const int half = (lane >> 4) & 1;
  const unsigned short* pa = Xbf + (size_t)row * D_DIM + half * 16;
  const v16bf* BP = reinterpret_cast<const v16bf*>(WPslot) + lane;

  v8us lo = *reinterpret_cast<const v8us*>(pa);        // K +0..7   (16B)
  v8us hi = *reinterpret_cast<const v8us*>(pa + 8);    // K +8..15  (16B)
  v16bf a = __builtin_bit_cast(v16bf,
      __builtin_shufflevector(lo, hi, 0, 1, 2, 3, 4, 5, 6, 7,
                                      8, 9, 10, 11, 12, 13, 14, 15));
  v16bf bcur = BP[0];
  v8us lo_n = lo, hi_n = hi;

#pragma unroll
  for (int i = 0; i < 128; ++i) {                      // i = kc*16 + nt
    const int kc = i >> 4;
    const int nt = i & 15;
    v16bf bnext = bcur;
    if (i < 127) bnext = BP[(i + 1) * 32];             // B double-buffer
    if (nt == 0 && kc < 7) {                           // A one chunk ahead
      const unsigned short* pn = pa + (kc + 1) * 32;
      lo_n = *reinterpret_cast<const v8us*>(pn);
      hi_n = *reinterpret_cast<const v8us*>(pn + 8);
    }
    acc[nt] = __builtin_amdgcn_wmma_f32_16x16x32_bf16(
        /*neg_a=*/false, a, /*neg_b=*/false, bcur,
        /*c_mod=*/(short)0, acc[nt], /*reuse_a=*/false, /*reuse_b=*/false);
    bcur = bnext;
    if (nt == 15)
      a = __builtin_bit_cast(v16bf,
          __builtin_shufflevector(lo_n, hi_n, 0, 1, 2, 3, 4, 5, 6, 7,
                                              8, 9, 10, 11, 12, 13, 14, 15));
  }
}

// C/D layout: VGPR r, lanes 0-15: M=r; lanes 16-31: M=8+r; N = (lane&15).
__device__ __forceinline__ size_t out_off(int m0, int lane, int nt, int r) {
  int m = m0 * 16 + 8 * ((lane >> 4) & 1) + r;
  int n = nt * 16 + (lane & 15);
  return (size_t)m * D_DIM + n;
}

// ---------------------------------------------------------------------------
// Kernel 1: fused 3-GEMM + activations (bias folded into epilogues).
//   inp  = (1-lam) * silu(x@Wi+bi);  lam = lb+(1-lb)*sigmoid(x@Wl+bl);
//   gate = silu(x@Wg+bg)
// ---------------------------------------------------------------------------
__global__ void __launch_bounds__(256, 1)
k_gemm3_act(const unsigned short* __restrict__ xbf, const unsigned short* __restrict__ wpack,
            const float* __restrict__ bi, const float* __restrict__ bl,
            const float* __restrict__ bg, const int* __restrict__ lbp,
            float* __restrict__ gate, float* __restrict__ lam,
            float* __restrict__ inp) {
  const int lane = threadIdx.x & 31;
  const int m0   = blockIdx.x * (blockDim.x >> 5) + (threadIdx.x >> 5);
  const int c0   = lane & 15;
  const float lb = (float)lbp[0];
  v8f acc[16];

  // pass 1: silu(x@Wi+bi) -> inp (temporary)
  wave_gemm16(xbf, wpack + 0 * WPACK_ELEMS, m0, lane, acc);
#pragma unroll
  for (int nt = 0; nt < 16; ++nt) {
    float bv = bi[nt * 16 + c0];
#pragma unroll
    for (int r = 0; r < 8; ++r) {
      float v = acc[nt][r] + bv;
      inp[out_off(m0, lane, nt, r)] = v * sig_(v);
    }
  }

  // pass 2: lam; combine inp = (1-lam)*inp
  wave_gemm16(xbf, wpack + 1 * WPACK_ELEMS, m0, lane, acc);
#pragma unroll
  for (int nt = 0; nt < 16; ++nt) {
    float bv = bl[nt * 16 + c0];
#pragma unroll
    for (int r = 0; r < 8; ++r) {
      size_t o = out_off(m0, lane, nt, r);
      float l = lb + (1.0f - lb) * sig_(acc[nt][r] + bv);
      lam[o] = l;
      inp[o] = (1.0f - l) * inp[o];    // same-thread RAW on global: kept in order
    }
  }

  // pass 3: gate = silu(x@Wg+bg)
  wave_gemm16(xbf, wpack + 2 * WPACK_ELEMS, m0, lane, acc);
#pragma unroll
  for (int nt = 0; nt < 16; ++nt) {
    float bv = bg[nt * 16 + c0];
#pragma unroll
    for (int r = 0; r < 8; ++r) {
      float v = acc[nt][r] + bv;
      gate[out_off(m0, lane, nt, r)] = v * sig_(v);
    }
  }
}

// ---------------------------------------------------------------------------
// Kernel 2: bidirectional scan along H (axis 0). Thread per (w,b, d/4).
//   hid = fwd + rev  (first writer). float4 over d.
// ---------------------------------------------------------------------------
__global__ void k_scan_h(const float* __restrict__ lam, const float* __restrict__ inp,
                         float* __restrict__ hid) {
  const size_t base   = ((size_t)blockIdx.x * blockDim.x + threadIdx.x) * 4; // over W*B*D
  const size_t stride = (size_t)W_DIM * B_DIM * D_DIM;                       // 262144
  float4 h = make_float4(0.f, 0.f, 0.f, 0.f);
  for (int t = 0; t < H_DIM; ++t) {
    size_t o = base + (size_t)t * stride;
    float4 l = *reinterpret_cast<const float4*>(lam + o);
    float4 i = *reinterpret_cast<const float4*>(inp + o);
    h.x = l.x * h.x + i.x; h.y = l.y * h.y + i.y;
    h.z = l.z * h.z + i.z; h.w = l.w * h.w + i.w;
    *reinterpret_cast<float4*>(hid + o) = h;
  }
  float4 r = make_float4(0.f, 0.f, 0.f, 0.f);
  for (int t = H_DIM - 1; t >= 0; --t) {
    size_t o = base + (size_t)t * stride;
    float4 l = *reinterpret_cast<const float4*>(lam + o);
    float4 i = *reinterpret_cast<const float4*>(inp + o);
    r.x = l.x * r.x + i.x; r.y = l.y * r.y + i.y;
    r.z = l.z * r.z + i.z; r.w = l.w * r.w + i.w;
    float4 c = *reinterpret_cast<const float4*>(hid + o);
    c.x += r.x; c.y += r.y; c.z += r.z; c.w += r.w;
    *reinterpret_cast<float4*>(hid + o) = c;
  }
}

// ---------------------------------------------------------------------------
// Kernel 3: bidirectional scan along W (axis 1). Thread per (h, (b,d)/4). hid += ...
// ---------------------------------------------------------------------------
__global__ void k_scan_w(const float* __restrict__ lam, const float* __restrict__ inp,
                         float* __restrict__ hid) {
  const int idx  = blockIdx.x * blockDim.x + threadIdx.x;      // 0 .. H*B*D/4-1
  const int hrow = idx >> 10;                                  // / (B*D/4 = 1024)
  const int rem  = idx & 1023;
  const size_t base   = (size_t)hrow * ((size_t)W_DIM * B_DIM * D_DIM) + (size_t)rem * 4;
  const size_t stride = (size_t)B_DIM * D_DIM;                 // 4096
  float4 h = make_float4(0.f, 0.f, 0.f, 0.f);
  for (int t = 0; t < W_DIM; ++t) {
    size_t o = base + (size_t)t * stride;
    float4 l = *reinterpret_cast<const float4*>(lam + o);
    float4 i = *reinterpret_cast<const float4*>(inp + o);
    h.x = l.x * h.x + i.x; h.y = l.y * h.y + i.y;
    h.z = l.z * h.z + i.z; h.w = l.w * h.w + i.w;
    float4 c = *reinterpret_cast<const float4*>(hid + o);
    c.x += h.x; c.y += h.y; c.z += h.z; c.w += h.w;
    *reinterpret_cast<float4*>(hid + o) = c;
  }
  float4 r = make_float4(0.f, 0.f, 0.f, 0.f);
  for (int t = W_DIM - 1; t >= 0; --t) {
    size_t o = base + (size_t)t * stride;
    float4 l = *reinterpret_cast<const float4*>(lam + o);
    float4 i = *reinterpret_cast<const float4*>(inp + o);
    r.x = l.x * r.x + i.x; r.y = l.y * r.y + i.y;
    r.z = l.z * r.z + i.z; r.w = l.w * r.w + i.w;
    float4 c = *reinterpret_cast<const float4*>(hid + o);
    c.x += r.x; c.y += r.y; c.z += r.z; c.w += r.w;
    *reinterpret_cast<float4*>(hid + o) = c;
  }
}

// ---------------------------------------------------------------------------
// Kernel 4: LayerNorm over D=256 per row, * gamma + beta, * gate -> feat (bf16)
// feat is consumed only as the A operand of the final WMMA GEMM.
// ---------------------------------------------------------------------------
__global__ void k_ln_gate(const float* __restrict__ hid, const float* __restrict__ gate,
                          const float* __restrict__ gamma, const float* __restrict__ beta,
                          unsigned short* __restrict__ featbf) {
  __shared__ float red[256];
  const int row = blockIdx.x;
  const int d   = threadIdx.x;
  const size_t o = (size_t)row * D_DIM + d;
  float v = hid[o];
  red[d] = v;
  __syncthreads();
#pragma unroll
  for (int s = 128; s > 0; s >>= 1) {
    if (d < s) red[d] += red[d + s];
    __syncthreads();
  }
  float mu = red[0] * (1.0f / 256.0f);
  __syncthreads();
  float c = v - mu;
  red[d] = c * c;
  __syncthreads();
#pragma unroll
  for (int s = 128; s > 0; s >>= 1) {
    if (d < s) red[d] += red[d + s];
    __syncthreads();
  }
  float var = red[0] * (1.0f / 256.0f);
  float f = c * rsqrtf(var + 1e-5f) * gamma[d] + beta[d];
  featbf[o] = f2bf(f * gate[o]);
}

// ---------------------------------------------------------------------------
// Kernel 5: out = feat @ Wo + bo
// ---------------------------------------------------------------------------
__global__ void __launch_bounds__(256, 1)
k_gemm_out(const unsigned short* __restrict__ featbf, const unsigned short* __restrict__ wpack,
           const float* __restrict__ bo, float* __restrict__ out) {
  const int lane = threadIdx.x & 31;
  const int m0   = blockIdx.x * (blockDim.x >> 5) + (threadIdx.x >> 5);
  const int c0   = lane & 15;
  v8f acc[16];
  wave_gemm16(featbf, wpack + 3 * WPACK_ELEMS, m0, lane, acc);
#pragma unroll
  for (int nt = 0; nt < 16; ++nt) {
    float bv = bo[nt * 16 + c0];
#pragma unroll
    for (int r = 0; r < 8; ++r)
      out[out_off(m0, lane, nt, r)] = acc[nt][r] + bv;
  }
}

// ---------------------------------------------------------------------------
extern "C" void kernel_launch(void* const* d_in, const int* in_sizes, int n_in,
                              void* d_out, int out_size, void* d_ws, size_t ws_size,
                              hipStream_t stream) {
  const float* x     = (const float*)d_in[0];
  const float* Wi    = (const float*)d_in[1];
  const float* Wl    = (const float*)d_in[2];
  const float* Wg    = (const float*)d_in[3];
  const float* Wo    = (const float*)d_in[4];
  const float* bi    = (const float*)d_in[5];
  const float* bl    = (const float*)d_in[6];
  const float* bg    = (const float*)d_in[7];
  const float* bo    = (const float*)d_in[8];
  const float* gamma = (const float*)d_in[9];
  const float* beta  = (const float*)d_in[10];
  const int*   lbp   = (const int*)d_in[11];

  // ws layout: [wpack bf16 512KB][xbf bf16 32MB][gate][lam][inp][hid] fp32
  // featbf reuses the xbf region (x dead after gemm3).
  unsigned short* wpack = (unsigned short*)d_ws;
  unsigned short* xbf   = wpack + 4 * WPACK_ELEMS;
  float* gate = (float*)(xbf + NTOT);
  float* lam  = gate + NTOT;
  float* inp  = lam + NTOT;
  float* hid  = inp + NTOT;
  unsigned short* featbf = xbf;  // reuse

  k_pack_weights<<<(4 * WPACK_ELEMS) / 256, 256, 0, stream>>>(Wi, Wl, Wg, Wo, wpack);
  k_cvt_x<<<(int)(NTOT / 8 / 256), 256, 0, stream>>>(x, xbf);
  k_gemm3_act<<<ROWS / 16 / 8, 256, 0, stream>>>(xbf, wpack, bi, bl, bg, lbp, gate, lam, inp);
  k_scan_h<<<(W_DIM * B_DIM * D_DIM / 4) / 256, 256, 0, stream>>>(lam, inp, hid);
  k_scan_w<<<(H_DIM * B_DIM * D_DIM / 4) / 256, 256, 0, stream>>>(lam, inp, hid);
  k_ln_gate<<<ROWS, 256, 0, stream>>>(hid, gate, gamma, beta, featbf);
  k_gemm_out<<<ROWS / 16 / 8, 256, 0, stream>>>(featbf, wpack, bo, (float*)d_out);
}
